// TinyLSTM_13623636263552
// MI455X (gfx1250) — matrix operand order repaired and checked by reading, not compiled, not run on hardware
//
#include <hip/hip_runtime.h>
#include <math.h>

// ---------------------------------------------------------------------------
// TinyLSTM on MI455X (gfx1250): bf16 WMMA GEMMs + persistent recurrent kernel
// ---------------------------------------------------------------------------
// Sizes (from reference): B=32, T=512, IN=512, H=1024, 4H=4096, 2 layers.

typedef unsigned short u16;
typedef __attribute__((ext_vector_type(16))) __bf16 v16bf;
typedef __attribute__((ext_vector_type(8)))  __bf16 v8bf;
typedef __attribute__((ext_vector_type(8)))  float  v8f;

union FragAB { v16bf v; v8bf h[2]; };

__device__ __forceinline__ u16 f2bf(float f) {
  unsigned u = __float_as_uint(f);
  unsigned r = 0x7FFFu + ((u >> 16) & 1u);
  return (u16)((u + r) >> 16);
}

__device__ __forceinline__ float sigmoidf_fast(float x) {
  return 1.0f / (1.0f + __expf(-x));
}

// Grid-wide barrier: monotonic counter, one increment per block per use.
__device__ __forceinline__ void grid_sync(unsigned* cnt, unsigned target) {
  __threadfence();                 // release all our global writes
  __syncthreads();
  if (threadIdx.x == 0) {
    __hip_atomic_fetch_add(cnt, 1u, __ATOMIC_ACQ_REL, __HIP_MEMORY_SCOPE_AGENT);
    while (__hip_atomic_load(cnt, __ATOMIC_ACQUIRE, __HIP_MEMORY_SCOPE_AGENT) < target) {
      __builtin_amdgcn_s_sleep(2);
    }
  }
  __syncthreads();
  __threadfence();                 // acquire other blocks' writes
}

// --------------------------- small utility kernels -------------------------

__global__ void f32_to_bf16_kernel(const float* __restrict__ in, u16* __restrict__ out, int n) {
  int i = blockIdx.x * blockDim.x + threadIdx.x;
  if (i < n) out[i] = f2bf(in[i]);
}

// Pack W [K, 4096] f32 row-major into WMMA B-fragment-linear bf16 layout:
// out[((nt*nkt + kt)*32 + lane)*16 + e] = bf16( W[kt*32 + (lane>=16?8:0) + (e&7) + (e>=8?16:0), nt*16 + (lane&15)] )
__global__ void pack_b_kernel(const float* __restrict__ W, u16* __restrict__ out, int K) {
  int idx = blockIdx.x * blockDim.x + threadIdx.x;
  int total = K * 4096;
  if (idx >= total) return;
  int e    = idx & 15;
  int lane = (idx >> 4) & 31;
  int tile = idx >> 9;
  int nkt  = K >> 5;
  int nt   = tile / nkt;
  int kt   = tile - nt * nkt;
  int n = nt * 16 + (lane & 15);
  int k = kt * 32 + ((lane >> 4) << 3) + (e & 7) + ((e >> 3) << 4);
  out[idx] = f2bf(W[(size_t)k * 4096 + n]);
}

__global__ void reset_sync_kernel(unsigned* cnt) { *cnt = 0u; }

// --------------------------- big input-projection GEMM ---------------------
// C[M,4096] = A[M,K](bf16,row-major) x Bpacked + bias.  One 16x16 tile / wave.

__global__ void __launch_bounds__(256) gemm_gi_kernel(
    const u16* __restrict__ A, const u16* __restrict__ Bp,
    const float* __restrict__ bias, float* __restrict__ C,
    int M, int K)
{
  const int lane   = threadIdx.x & 31;
  const int lanelo = lane & 15;
  const int hi     = lane >> 4;
  const int wave   = blockIdx.x * 8 + (threadIdx.x >> 5);
  const int mt     = wave >> 8;     // / 256 N-tiles
  const int nt     = wave & 255;
  if (mt >= (M >> 4)) return;       // wave-uniform
  const int nkt = K >> 5;

  v8f acc0 = {0.f,0.f,0.f,0.f,0.f,0.f,0.f,0.f};
  v8f acc1 = {0.f,0.f,0.f,0.f,0.f,0.f,0.f,0.f};

  const u16* arow  = A + (size_t)(mt * 16 + lanelo) * K + hi * 8;
  const u16* bbase = Bp + ((size_t)nt * nkt * 32 + lane) * 16;

  for (int kt = 0; kt < nkt; kt += 2) {
    FragAB a0, b0, a1, b1;
    const u16* ap0 = arow + kt * 32;
    a0.h[0] = *(const v8bf*)(ap0);
    a0.h[1] = *(const v8bf*)(ap0 + 16);
    b0.v    = *(const v16bf*)(bbase + (size_t)kt * 512);
    acc0 = __builtin_amdgcn_wmma_f32_16x16x32_bf16(false, a0.v, false, b0.v,
                                                   (short)0, acc0, false, false);
    const u16* ap1 = arow + (kt + 1) * 32;
    a1.h[0] = *(const v8bf*)(ap1);
    a1.h[1] = *(const v8bf*)(ap1 + 16);
    b1.v    = *(const v16bf*)(bbase + (size_t)(kt + 1) * 512);
    acc1 = __builtin_amdgcn_wmma_f32_16x16x32_bf16(false, a1.v, false, b1.v,
                                                   (short)0, acc1, false, false);
  }

  const int col = nt * 16 + lanelo;
  const float bv = bias[col];
  const int rowbase = mt * 16 + hi * 8;
  #pragma unroll
  for (int r = 0; r < 8; ++r)
    C[(size_t)(rowbase + r) * 4096 + col] = acc0[r] + acc1[r] + bv;
}

// --------------------------- persistent LSTM recurrence --------------------
// 64 blocks x 256 threads = 512 wave32s = exactly the 2x256 output tiles of
// the per-step [32,1024]x[1024,4096] GEMM. Two grid barriers per timestep.

#define NBLK 64
#define NTHR 256
#define NALL (NBLK * NTHR)
#define SEQLEN 512

__global__ void __launch_bounds__(256) lstm_recurrent_kernel(
    const float* __restrict__ gi,   // [(b*SEQ+t)*4096]  (x@w_ih + b_ih)
    const u16*   __restrict__ Whp,  // packed bf16 w_hh fragments (nkt=32)
    const float* __restrict__ bhh,  // [4096]
    float* __restrict__ hstate,     // [32*1024]
    float* __restrict__ cstate,     // [32*1024]
    u16*   __restrict__ hbf,        // [32*1024] bf16 h for next-step A operand
    float* __restrict__ gates,      // [32*4096] post-sigmoid gates
    u16*   __restrict__ hs_bf,      // [(b*SEQ+t)*1024] bf16 seq out (next layer in)
    float* __restrict__ hs_f32,     // optional f32 seq out (final layer) or null
    float* __restrict__ hT,         // [32*1024]
    float* __restrict__ cT,         // [32*1024]
    unsigned* __restrict__ sync)
{
  const int lane   = threadIdx.x & 31;
  const int lanelo = lane & 15;
  const int hi     = lane >> 4;
  const int wave   = blockIdx.x * 8 + (threadIdx.x >> 5);
  const int mt     = wave >> 8;
  const int nt     = wave & 255;
  const int gtid   = blockIdx.x * NTHR + threadIdx.x;
  unsigned bar = 0;

  // zero initial state (h0 = c0 = 0)
  for (int i = gtid; i < 32 * 1024; i += NALL) {
    hstate[i] = 0.f; cstate[i] = 0.f; hbf[i] = 0;
  }
  grid_sync(sync, ++bar * NBLK);

  const u16* bbase = Whp + ((size_t)nt * 32 * 32 + lane) * 16;   // nkt = 32
  const int  col     = nt * 16 + lanelo;
  const float bv     = bhh[col];
  const int  rowbase = mt * 16 + hi * 8;
  const u16* arow    = hbf + (size_t)(mt * 16 + lanelo) * 1024 + hi * 8;

  for (int t = 0; t < SEQLEN; ++t) {
    // ---- Phase A: gates = sigmoid(gi_t + h @ w_hh + b_hh) -----------------
    v8f acc0, acc1;
    #pragma unroll
    for (int r = 0; r < 8; ++r) {
      acc0[r] = gi[((size_t)(rowbase + r) * SEQLEN + t) * 4096 + col] + bv;
      acc1[r] = 0.f;
    }
    #pragma unroll 4
    for (int kt = 0; kt < 32; kt += 2) {
      FragAB a0, b0, a1, b1;
      const u16* ap0 = arow + kt * 32;
      a0.h[0] = *(const v8bf*)(ap0);
      a0.h[1] = *(const v8bf*)(ap0 + 16);
      b0.v    = *(const v16bf*)(bbase + (size_t)kt * 512);
      acc0 = __builtin_amdgcn_wmma_f32_16x16x32_bf16(false, a0.v, false, b0.v,
                                                     (short)0, acc0, false, false);
      const u16* ap1 = arow + (kt + 1) * 32;
      a1.h[0] = *(const v8bf*)(ap1);
      a1.h[1] = *(const v8bf*)(ap1 + 16);
      b1.v    = *(const v16bf*)(bbase + (size_t)(kt + 1) * 512);
      acc1 = __builtin_amdgcn_wmma_f32_16x16x32_bf16(false, a1.v, false, b1.v,
                                                     (short)0, acc1, false, false);
    }
    #pragma unroll
    for (int r = 0; r < 8; ++r) {
      float g = sigmoidf_fast(acc0[r] + acc1[r]);
      gates[(size_t)(rowbase + r) * 4096 + col] = g;
    }
    grid_sync(sync, ++bar * NBLK);

    // ---- Phase B: c,h update (all four gates are sigmoid per reference) ---
    for (int i = gtid; i < 32 * 1024; i += NALL) {
      int b = i >> 10, j = i & 1023;
      const float* gb = gates + (size_t)b * 4096;
      float fg = gb[j];
      float ig = gb[1024 + j];
      float ng = gb[2048 + j];
      float og = gb[3072 + j];
      float c  = fg * cstate[i] + ig * ng;
      cstate[i] = c;
      float h  = og * tanhf(c);
      hstate[i] = h;
      u16 hb = f2bf(h);
      hbf[i] = hb;
      size_t orow = ((size_t)b * SEQLEN + t) * 1024 + j;
      hs_bf[orow] = hb;
      if (hs_f32) hs_f32[orow] = h;
    }
    grid_sync(sync, ++bar * NBLK);
  }

  // final h/c
  for (int i = gtid; i < 32 * 1024; i += NALL) {
    hT[i] = hstate[i];
    cT[i] = cstate[i];
  }
}

// --------------------------------- launch ----------------------------------

extern "C" void kernel_launch(void* const* d_in, const int* in_sizes, int n_in,
                              void* d_out, int out_size, void* d_ws, size_t ws_size,
                              hipStream_t stream) {
  (void)in_sizes; (void)n_in; (void)out_size; (void)ws_size;
  const float* x    = (const float*)d_in[0];
  const float* wih0 = (const float*)d_in[1];
  const float* whh0 = (const float*)d_in[2];
  const float* bih0 = (const float*)d_in[3];
  const float* bhh0 = (const float*)d_in[4];
  const float* wih1 = (const float*)d_in[5];
  const float* whh1 = (const float*)d_in[6];
  const float* bih1 = (const float*)d_in[7];
  const float* bhh1 = (const float*)d_in[8];
  float* out = (float*)d_out;

  char* ws = (char*)d_ws;
  size_t off = 0;
  auto alloc = [&](size_t bytes) -> void* {
    void* p = ws + off;
    off += (bytes + 255) & ~(size_t)255;
    return p;
  };

  float*    gi     = (float*)alloc((size_t)16384 * 4096 * 4); // shared by both layers
  u16*      hs0bf  = (u16*)  alloc((size_t)16384 * 1024 * 2);
  u16*      xbf    = (u16*)  alloc((size_t)16384 * 512 * 2);
  u16*      wih0p  = (u16*)  alloc((size_t)512  * 4096 * 2);
  u16*      whh0p  = (u16*)  alloc((size_t)1024 * 4096 * 2);
  u16*      wih1p  = (u16*)  alloc((size_t)1024 * 4096 * 2);
  u16*      whh1p  = (u16*)  alloc((size_t)1024 * 4096 * 2);
  float*    hstate = (float*)alloc((size_t)32 * 1024 * 4);
  float*    cstate = (float*)alloc((size_t)32 * 1024 * 4);
  u16*      hbf    = (u16*)  alloc((size_t)32 * 1024 * 2);
  float*    gates  = (float*)alloc((size_t)32 * 4096 * 4);
  unsigned* syncc  = (unsigned*)alloc(256);

  const size_t SEQ_ELEMS = (size_t)32 * 512 * 1024;
  float* seq_out = out;
  float* hT0 = out + SEQ_ELEMS;
  float* hT1 = out + SEQ_ELEMS + 32 * 1024;
  float* cT0 = out + SEQ_ELEMS + 2 * 32 * 1024;
  float* cT1 = out + SEQ_ELEMS + 3 * 32 * 1024;

  // 1) convert x to bf16, pack all weights into WMMA fragment layout
  f32_to_bf16_kernel<<<(16384 * 512 + 255) / 256, 256, 0, stream>>>(x, xbf, 16384 * 512);
  pack_b_kernel<<<(512  * 4096 + 255) / 256, 256, 0, stream>>>(wih0, wih0p, 512);
  pack_b_kernel<<<(1024 * 4096 + 255) / 256, 256, 0, stream>>>(whh0, whh0p, 1024);
  pack_b_kernel<<<(1024 * 4096 + 255) / 256, 256, 0, stream>>>(wih1, wih1p, 1024);
  pack_b_kernel<<<(1024 * 4096 + 255) / 256, 256, 0, stream>>>(whh1, whh1p, 1024);

  // 2) layer 0: input projection GEMM, then persistent recurrence
  gemm_gi_kernel<<<32768, 256, 0, stream>>>(xbf, wih0p, bih0, gi, 16384, 512);
  reset_sync_kernel<<<1, 1, 0, stream>>>(syncc);
  lstm_recurrent_kernel<<<NBLK, NTHR, 0, stream>>>(gi, whh0p, bhh0, hstate, cstate,
                                                   hbf, gates, hs0bf, nullptr,
                                                   hT0, cT0, syncc);

  // 3) layer 1: input projection GEMM over layer-0 hidden seq, then recurrence
  gemm_gi_kernel<<<32768, 256, 0, stream>>>(hs0bf, wih1p, bih1, gi, 16384, 1024);
  reset_sync_kernel<<<1, 1, 0, stream>>>(syncc);
  lstm_recurrent_kernel<<<NBLK, NTHR, 0, stream>>>(gi, whh1p, bhh1, hstate, cstate,
                                                   hbf, gates, hs0bf /*scratch*/, seq_out,
                                                   hT1, cT1, syncc);
}
